// NNCLR_7782480740655
// MI455X (gfx1250) — matrix operand order repaired and checked by reading, not compiled, not run on hardware
//
#include <hip/hip_runtime.h>
#include <hip/hip_fp16.h>

typedef __attribute__((ext_vector_type(16))) _Float16 v16h;
typedef __attribute__((ext_vector_type(8)))  _Float16 v8h;
typedef __attribute__((ext_vector_type(4)))  _Float16 v4h;
typedef __attribute__((ext_vector_type(8)))  float    v8f;

#define DDIM 128
#define TEMP_INV 10.0f   // 1 / TEMPERATURE (also the safe fixed LSE shift: |logit| <= ~10)

// ---------------------------------------------------------------------------
// WMMA fragment loaders (16x16x32 f16, wave32; contiguous k order shared by A and B, not the ISA table order)
// A: 16x32 (MxK). lane l: M = l%16, hi = l/16.
//    elem e (0..15): K = kt + 16*hi + e                 -> one contiguous v16h.
// B: 32x16 (KxN) with B[k][n] = Bmem[n][k] (row-major [N,128]).
//    lane l: N = l%16, elem e: K = kt + 16*hi + e       -> one contiguous v16h.
// ---------------------------------------------------------------------------
__device__ __forceinline__ v16h load_frag_a(const _Float16* A, int m0, int kt) {
    const int l  = threadIdx.x & 31;
    const int hi = l >> 4;
    return *(const v16h*)(A + (m0 + (l & 15)) * DDIM + kt + 16 * hi);
}

__device__ __forceinline__ v16h load_frag_b(const _Float16* Bm, int n0, int kt) {
    const int l  = threadIdx.x & 31;
    const int hi = l >> 4;
    return *(const v16h*)(Bm + (n0 + (l & 15)) * DDIM + kt + 16 * hi);
}

// ---------------------------------------------------------------------------
// Retrieval: each block owns 32 rows (two 16-row strips per wave, so every
// streamed queue tile feeds 8 WMMAs). Streaming argmax over all Q columns.
// C/D layout: reg r of lane l holds (row = r + 8*(l/16), col = n0 + l%16).
// ---------------------------------------------------------------------------
__global__ void __launch_bounds__(256) argmax_rows(const _Float16* __restrict__ Ph,
                                                   const _Float16* __restrict__ Qh,
                                                   int* __restrict__ idx_out, int Q) {
    const int m0   = blockIdx.x * 32;
    const int lane = threadIdx.x & 31;
    const int wv   = threadIdx.x >> 5;
    const int nw   = blockDim.x >> 5;

    v16h a[8];
#pragma unroll
    for (int k = 0; k < 4; ++k) {
        a[k]     = load_frag_a(Ph, m0,      k * 32);
        a[4 + k] = load_frag_a(Ph, m0 + 16, k * 32);
    }

    float best0[8], best1[8]; int bidx0[8], bidx1[8];
#pragma unroll
    for (int r = 0; r < 8; ++r) {
        best0[r] = -3.0e38f; best1[r] = -3.0e38f;
        bidx0[r] = 0x7fffffff; bidx1[r] = 0x7fffffff;
    }

    const int ntiles = Q >> 4;
    for (int t = wv; t < ntiles; t += nw) {
        const int n0 = t << 4;
        v16h b0 = load_frag_b(Qh, n0, 0);
        v16h b1 = load_frag_b(Qh, n0, 32);
        v16h b2 = load_frag_b(Qh, n0, 64);
        v16h b3 = load_frag_b(Qh, n0, 96);
        v8f c0 = {};
        c0 = __builtin_amdgcn_wmma_f32_16x16x32_f16(false, a[0], false, b0, (short)0, c0, false, false);
        c0 = __builtin_amdgcn_wmma_f32_16x16x32_f16(false, a[1], false, b1, (short)0, c0, false, false);
        c0 = __builtin_amdgcn_wmma_f32_16x16x32_f16(false, a[2], false, b2, (short)0, c0, false, false);
        c0 = __builtin_amdgcn_wmma_f32_16x16x32_f16(false, a[3], false, b3, (short)0, c0, false, false);
        v8f c1 = {};
        c1 = __builtin_amdgcn_wmma_f32_16x16x32_f16(false, a[4], false, b0, (short)0, c1, false, false);
        c1 = __builtin_amdgcn_wmma_f32_16x16x32_f16(false, a[5], false, b1, (short)0, c1, false, false);
        c1 = __builtin_amdgcn_wmma_f32_16x16x32_f16(false, a[6], false, b2, (short)0, c1, false, false);
        c1 = __builtin_amdgcn_wmma_f32_16x16x32_f16(false, a[7], false, b3, (short)0, c1, false, false);
        const int col = n0 + (lane & 15);
        // strict '>' keeps the first (lowest-column) max within a lane's stream
#pragma unroll
        for (int r = 0; r < 8; ++r) {
            if (c0[r] > best0[r]) { best0[r] = c0[r]; bidx0[r] = col; }
            if (c1[r] > best1[r]) { best1[r] = c1[r]; bidx1[r] = col; }
        }
    }
    // reduce across the 16 lanes of each half-wave (same rows, different cols)
#pragma unroll
    for (int off = 1; off < 16; off <<= 1) {
#pragma unroll
        for (int r = 0; r < 8; ++r) {
            float ov = __shfl_xor(best0[r], off, 32);
            int   oi = __shfl_xor(bidx0[r], off, 32);
            if (ov > best0[r] || (ov == best0[r] && oi < bidx0[r])) { best0[r] = ov; bidx0[r] = oi; }
            ov = __shfl_xor(best1[r], off, 32);
            oi = __shfl_xor(bidx1[r], off, 32);
            if (ov > best1[r] || (ov == best1[r] && oi < bidx1[r])) { best1[r] = ov; bidx1[r] = oi; }
        }
    }
    __shared__ float sv[8][32];
    __shared__ int   si[8][32];
    if (lane == 0 || lane == 16) {
        const int rb = (lane >> 4) * 8;
#pragma unroll
        for (int r = 0; r < 8; ++r) {
            sv[wv][rb + r]      = best0[r]; si[wv][rb + r]      = bidx0[r];
            sv[wv][16 + rb + r] = best1[r]; si[wv][16 + rb + r] = bidx1[r];
        }
    }
    __syncthreads();
    if (threadIdx.x < 32) {
        float bv = sv[0][threadIdx.x]; int bi = si[0][threadIdx.x];
        for (int w = 1; w < nw; ++w) {
            const float ov = sv[w][threadIdx.x]; const int oi = si[w][threadIdx.x];
            if (ov > bv || (ov == bv && oi < bi)) { bv = ov; bi = oi; }
        }
        idx_out[m0 + threadIdx.x] = bi;
    }
}

// ---------------------------------------------------------------------------
// loss[i] = logsumexp_j(A[i].B[j]*10) - diag[i]*10.
// All logits are bounded by ~10 (unit vectors, 1/t = 10), so a FIXED shift of
// 10 is numerically safe: lse = 10 + log(sum exp(x-10)). No online max needed.
// Two 16-row strips per wave (32 rows/block) for B-tile reuse.
// ---------------------------------------------------------------------------
__global__ void __launch_bounds__(256) lse_loss_rows(const _Float16* __restrict__ Ah,
                                                     const _Float16* __restrict__ Bh,
                                                     const float* __restrict__ diag,
                                                     float* __restrict__ loss_out, int N) {
    const int m0   = blockIdx.x * 32;
    const int lane = threadIdx.x & 31;
    const int wv   = threadIdx.x >> 5;
    const int nw   = blockDim.x >> 5;

    v16h a[8];
#pragma unroll
    for (int k = 0; k < 4; ++k) {
        a[k]     = load_frag_a(Ah, m0,      k * 32);
        a[4 + k] = load_frag_a(Ah, m0 + 16, k * 32);
    }

    float s0[8], s1[8];
#pragma unroll
    for (int r = 0; r < 8; ++r) { s0[r] = 0.0f; s1[r] = 0.0f; }

    const int ntiles = N >> 4;
    for (int t = wv; t < ntiles; t += nw) {
        const int n0 = t << 4;
        v16h b0 = load_frag_b(Bh, n0, 0);
        v16h b1 = load_frag_b(Bh, n0, 32);
        v16h b2 = load_frag_b(Bh, n0, 64);
        v16h b3 = load_frag_b(Bh, n0, 96);
        v8f c0 = {};
        c0 = __builtin_amdgcn_wmma_f32_16x16x32_f16(false, a[0], false, b0, (short)0, c0, false, false);
        c0 = __builtin_amdgcn_wmma_f32_16x16x32_f16(false, a[1], false, b1, (short)0, c0, false, false);
        c0 = __builtin_amdgcn_wmma_f32_16x16x32_f16(false, a[2], false, b2, (short)0, c0, false, false);
        c0 = __builtin_amdgcn_wmma_f32_16x16x32_f16(false, a[3], false, b3, (short)0, c0, false, false);
        v8f c1 = {};
        c1 = __builtin_amdgcn_wmma_f32_16x16x32_f16(false, a[4], false, b0, (short)0, c1, false, false);
        c1 = __builtin_amdgcn_wmma_f32_16x16x32_f16(false, a[5], false, b1, (short)0, c1, false, false);
        c1 = __builtin_amdgcn_wmma_f32_16x16x32_f16(false, a[6], false, b2, (short)0, c1, false, false);
        c1 = __builtin_amdgcn_wmma_f32_16x16x32_f16(false, a[7], false, b3, (short)0, c1, false, false);
#pragma unroll
        for (int r = 0; r < 8; ++r) {
            s0[r] += __expf(__builtin_fmaf(c0[r], TEMP_INV, -TEMP_INV));
            s1[r] += __expf(__builtin_fmaf(c1[r], TEMP_INV, -TEMP_INV));
        }
    }
    // plain sum across the 16 lanes of each half-wave
#pragma unroll
    for (int off = 1; off < 16; off <<= 1) {
#pragma unroll
        for (int r = 0; r < 8; ++r) {
            s0[r] += __shfl_xor(s0[r], off, 32);
            s1[r] += __shfl_xor(s1[r], off, 32);
        }
    }
    __shared__ float ss[8][32];
    if (lane == 0 || lane == 16) {
        const int rb = (lane >> 4) * 8;
#pragma unroll
        for (int r = 0; r < 8; ++r) {
            ss[wv][rb + r]      = s0[r];
            ss[wv][16 + rb + r] = s1[r];
        }
    }
    __syncthreads();
    if (threadIdx.x < 32) {
        float tot = ss[0][threadIdx.x];
        for (int w = 1; w < nw; ++w) tot += ss[w][threadIdx.x];
        loss_out[m0 + threadIdx.x] =
            TEMP_INV + logf(tot) - diag[m0 + threadIdx.x] * TEMP_INV;
    }
}

// ---------------------------------------------------------------------------
// Helpers: normalize rows (f32 out + f16 out), f32->f16, gather, diag dots
// ---------------------------------------------------------------------------
__global__ void __launch_bounds__(256) normalize_rows(const float* __restrict__ X,
                                                      float* __restrict__ Xn,
                                                      _Float16* __restrict__ Xh, int B) {
    const int row = blockIdx.x * (blockDim.x >> 5) + (threadIdx.x >> 5);
    if (row >= B) return;
    const int lane = threadIdx.x & 31;
    float4 v = *(const float4*)(X + (size_t)row * DDIM + lane * 4);
    float sumsq = v.x * v.x + v.y * v.y + v.z * v.z + v.w * v.w;
#pragma unroll
    for (int off = 16; off; off >>= 1) sumsq += __shfl_xor(sumsq, off, 32);
    const float rn = rsqrtf(sumsq);
    float4 o; o.x = v.x * rn; o.y = v.y * rn; o.z = v.z * rn; o.w = v.w * rn;
    *(float4*)(Xn + (size_t)row * DDIM + lane * 4) = o;
    v4h h; h[0] = (_Float16)o.x; h[1] = (_Float16)o.y; h[2] = (_Float16)o.z; h[3] = (_Float16)o.w;
    *(v4h*)(Xh + (size_t)row * DDIM + lane * 4) = h;
}

__global__ void __launch_bounds__(256) f32_to_f16(const float* __restrict__ src,
                                                  _Float16* __restrict__ dst, int n) {
    const int i = blockIdx.x * blockDim.x + threadIdx.x;
    if (i < n) dst[i] = (_Float16)src[i];
}

__global__ void __launch_bounds__(256) gather_rows_f16(const _Float16* __restrict__ Qh,
                                                       const int* __restrict__ idx,
                                                       _Float16* __restrict__ out, int n) {
    const int i = blockIdx.x * blockDim.x + threadIdx.x;
    if (i < n) {
        const int row = i >> 7;          // DDIM == 128
        const int k   = i & (DDIM - 1);
        out[i] = Qh[(size_t)idx[row] * DDIM + k];
    }
}

__global__ void __launch_bounds__(256) diag_dot(const float* __restrict__ Qf,
                                                const int* __restrict__ idx,
                                                const float* __restrict__ Pn,
                                                float* __restrict__ diag, int B) {
    const int row = blockIdx.x * (blockDim.x >> 5) + (threadIdx.x >> 5);
    if (row >= B) return;
    const int lane = threadIdx.x & 31;
    const float4 a = *(const float4*)(Qf + (size_t)idx[row] * DDIM + lane * 4);
    const float4 b = *(const float4*)(Pn + (size_t)row * DDIM + lane * 4);
    float d = a.x * b.x + a.y * b.y + a.z * b.z + a.w * b.w;
#pragma unroll
    for (int off = 16; off; off >>= 1) d += __shfl_xor(d, off, 32);
    if (lane == 0) diag[row] = d;
}

// ---------------------------------------------------------------------------
extern "C" void kernel_launch(void* const* d_in, const int* in_sizes, int n_in,
                              void* d_out, int out_size, void* d_ws, size_t ws_size,
                              hipStream_t stream) {
    const float* p1 = (const float*)d_in[0];
    const float* p2 = (const float*)d_in[1];
    const float* qf = (const float*)d_in[2];
    const int B = in_sizes[0] / DDIM;   // 4096
    const int Q = in_sizes[2] / DDIM;   // 32768
    float* loss = (float*)d_out;        // [4B]

    unsigned char* ws = (unsigned char*)d_ws;
    size_t off = 0;
    auto alloc = [&](size_t bytes) -> void* {
        void* p = ws + off;
        off += (bytes + 255) & ~(size_t)255;
        return p;
    };
    _Float16* p1h = (_Float16*)alloc((size_t)B * DDIM * 2);
    _Float16* p2h = (_Float16*)alloc((size_t)B * DDIM * 2);
    _Float16* qh  = (_Float16*)alloc((size_t)Q * DDIM * 2);
    _Float16* n1h = (_Float16*)alloc((size_t)B * DDIM * 2);
    _Float16* n2h = (_Float16*)alloc((size_t)B * DDIM * 2);
    float*    p1n = (float*)   alloc((size_t)B * DDIM * 4);
    float*    p2n = (float*)   alloc((size_t)B * DDIM * 4);
    int*      id1 = (int*)     alloc((size_t)B * 4);
    int*      id2 = (int*)     alloc((size_t)B * 4);
    float*    dg1 = (float*)   alloc((size_t)B * 4);
    float*    dg2 = (float*)   alloc((size_t)B * 4);

    // 1) normalize projections (f32 + f16 copies)
    normalize_rows<<<(B + 7) / 8, 256, 0, stream>>>(p1, p1n, p1h, B);
    normalize_rows<<<(B + 7) / 8, 256, 0, stream>>>(p2, p2n, p2h, B);

    // 2) queue f32 -> f16 (queue is already unit-norm)
    const int nq = Q * DDIM;
    f32_to_f16<<<(nq + 255) / 256, 256, 0, stream>>>(qf, qh, nq);

    // 3) top-1 retrieval via streamed WMMA argmax (32 rows per block)
    argmax_rows<<<B / 32, 256, 0, stream>>>(p1h, qh, id1, Q);
    argmax_rows<<<B / 32, 256, 0, stream>>>(p2h, qh, id2, Q);

    // 4) gather neighbours (f16 rows for the logit GEMMs)
    const int nb = B * DDIM;
    gather_rows_f16<<<(nb + 255) / 256, 256, 0, stream>>>(qh, id1, n1h, nb);
    gather_rows_f16<<<(nb + 255) / 256, 256, 0, stream>>>(qh, id2, n2h, nb);

    // 5) f32 diagonal dots: dg1[i] = n1[i].p2[i], dg2[i] = n2[i].p1[i]
    diag_dot<<<(B + 7) / 8, 256, 0, stream>>>(qf, id1, p2n, dg1, B);
    diag_dot<<<(B + 7) / 8, 256, 0, stream>>>(qf, id2, p1n, dg2, B);

    // 6) four streamed fixed-shift sum-exp blocks of the [4B,B] logits
    lse_loss_rows<<<B / 32, 256, 0, stream>>>(n1h, p2h, dg1, loss + 0 * B, B); // s_1_2_1
    lse_loss_rows<<<B / 32, 256, 0, stream>>>(p2h, n1h, dg1, loss + 1 * B, B); // s_1_2_2
    lse_loss_rows<<<B / 32, 256, 0, stream>>>(n2h, p1h, dg2, loss + 2 * B, B); // s_2_1_1
    lse_loss_rows<<<B / 32, 256, 0, stream>>>(p1h, n2h, dg2, loss + 3 * B, B); // s_2_1_2
}
